// MAB_74285754352438
// MI455X (gfx1250) — matrix operand order repaired and checked by reading, not compiled, not run on hardware
//
#include <hip/hip_runtime.h>
#include <math.h>

// ---------------------------------------------------------------- constants
constexpr int BB  = 32;
constexpr int NQ  = 1024;
constexpr int NK  = 1024;
constexpr int DV  = 512;
constexpr int H   = 8;
constexpr int DH  = 64;   // DV / H

typedef __attribute__((ext_vector_type(16))) _Float16 v16h;
typedef __attribute__((ext_vector_type(8)))  float    v8f;

// ---------------------------------------------------------------- WMMA operand loaders
// A-operand (16x32, M x K), 16-bit: lanes<16 hold K {k0..k0+7, k0+16..k0+23},
// lanes>=16 hold K {k0+8..k0+15, k0+24..k0+31}; lane&15 = M row.
static __device__ __forceinline__ v16h load_a_f16(const _Float16* __restrict__ src,
                                                  int ld, int row, int k0, int lane) {
  const int base = (lane & 16) ? 8 : 0;
  const _Float16* p0 = src + (size_t)row * ld + k0 + base;
  const _Float16* p1 = p0 + 16;
  v16h a;
#pragma unroll
  for (int i = 0; i < 8; ++i) { a[i] = p0[i]; a[i + 8] = p1[i]; }
  return a;
}

// B-operand (32x16, K x N), 16-bit: lane&15 = N column; lanes<16 hold K {k0..k0+7, k0+16..k0+23},
// lanes>=16 hold K {k0+8..k0+15, k0+24..k0+31}. Loaded from the row-major B^T (N x K) source.
static __device__ __forceinline__ v16h load_b_f16(const _Float16* __restrict__ src,
                                                  int ld, int nrow, int k0, int lane) {
  const _Float16* p = src + (size_t)nrow * ld + k0 + ((lane & 16) ? 8 : 0);
  v16h b;
#pragma unroll
  for (int i = 0; i < 8; ++i) { b[i] = p[i]; b[i + 8] = p[i + 16]; }
  return b;
}

// ---------------------------------------------------------------- kernel 0: f32 -> f16 convert
// One pass over the big operands so the GEMM hot loops carry zero v_cvt work.
__global__ __launch_bounds__(256)
void mab_cvt_f16(const float* __restrict__ X, _Float16* __restrict__ Y, size_t n) {
  const size_t i0 = ((size_t)blockIdx.x * blockDim.x + threadIdx.x) * 8;
  if (i0 + 8 > n) return;
  const float4 a = *(const float4*)(X + i0);
  const float4 b = *(const float4*)(X + i0 + 4);
  v16h o16;  // unused; the 8 halves are written as scalar f16 stores below
  _Float16 o[8];
  o[0] = (_Float16)a.x; o[1] = (_Float16)a.y; o[2] = (_Float16)a.z; o[3] = (_Float16)a.w;
  o[4] = (_Float16)b.x; o[5] = (_Float16)b.y; o[6] = (_Float16)b.z; o[7] = (_Float16)b.w;
  (void)o16;
#pragma unroll
  for (int i = 0; i < 8; ++i) Y[i0 + i] = o[i];
}

// ---------------------------------------------------------------- kernel 1: projection GEMM
// Wave tile = 32(M) x 64(N): 8 independent accumulators (no WMMA D-chain hazard
// bubbles), A operand reused 4x, B operand 2x; all-f16 loads (pre-converted).
__global__ __launch_bounds__(256)
void mab_proj_gemm(const _Float16* __restrict__ A, const _Float16* __restrict__ W,
                   const float* __restrict__ bias,
                   float* __restrict__ out32, _Float16* __restrict__ out16, int M) {
  const int lane = threadIdx.x & 31;
  const int wave = (blockIdx.x * blockDim.x + threadIdx.x) >> 5;
  const int ngrp = DV / 64;                  // 8 column groups
  const int tm   = (wave / ngrp) * 32;
  const int tng  = (wave % ngrp) * 64;
  if (tm >= M) return;
  const int l15 = lane & 15;

  v8f acc[2][4];
#pragma unroll
  for (int mt = 0; mt < 2; ++mt)
#pragma unroll
    for (int t = 0; t < 4; ++t) acc[mt][t] = (v8f){0.f,0.f,0.f,0.f,0.f,0.f,0.f,0.f};

  for (int k = 0; k < DV; k += 32) {
    v16h a0 = load_a_f16(A, DV, tm + l15,      k, lane);
    v16h a1 = load_a_f16(A, DV, tm + 16 + l15, k, lane);
    v16h bm[4];
#pragma unroll
    for (int t = 0; t < 4; ++t) bm[t] = load_b_f16(W, DV, tng + t * 16 + l15, k, lane);
#pragma unroll
    for (int t = 0; t < 4; ++t) {
      acc[0][t] = __builtin_amdgcn_wmma_f32_16x16x32_f16(false, a0, false, bm[t],
                                                         (short)0, acc[0][t], false, false);
      acc[1][t] = __builtin_amdgcn_wmma_f32_16x16x32_f16(false, a1, false, bm[t],
                                                         (short)0, acc[1][t], false, false);
    }
  }
  const int mofs = (lane & 16) ? 8 : 0;
#pragma unroll
  for (int mt = 0; mt < 2; ++mt)
#pragma unroll
    for (int t = 0; t < 4; ++t) {
      const int col = tng + t * 16 + l15;
      const float bv = bias[col];
#pragma unroll
      for (int r = 0; r < 8; ++r) {
        const float v   = acc[mt][t][r] + bv;
        const size_t ix = (size_t)(tm + mt * 16 + mofs + r) * DV + col;
        if (out32) out32[ix] = v;
        if (out16) out16[ix] = (_Float16)v;
      }
    }
}

// ---------------------------------------------------------------- kernel 2: V transpose
// Vp [b, nk, h*64+d] f16 -> Vt [b, h, d, nk] f16; index mapped so STORES are
// contiguous (reads stride 1KB and ride L2).
__global__ __launch_bounds__(256)
void mab_transpose_v(const _Float16* __restrict__ Vp, _Float16* __restrict__ Vt) {
  const size_t idx = (size_t)blockIdx.x * blockDim.x + threadIdx.x;
  const size_t total = (size_t)BB * NK * DV;
  if (idx >= total) return;
  const int nk   = (int)(idx % NK);
  const size_t t = idx / NK;
  const int v    = (int)(t % DV);
  const int b    = (int)(t / DV);
  const int h = v >> 6, d = v & 63;
  Vt[(((size_t)b * H + h) * DH + d) * NK + nk] = Vp[((size_t)b * NK + nk) * DV + v];
}

// ---------------------------------------------------------------- kernel 3: flash attention
// One wave owns (b, h, 32 query rows = 2 q-tiles); streams NK in 32-wide tiles
// with online softmax. K/V operand loads shared across both q-tiles; paired
// WMMAs are independent. S^T D-layout -> P A-operand pack is lane-local.
__global__ __launch_bounds__(256)
void mab_attention(const _Float16* __restrict__ Qp, const _Float16* __restrict__ Kp,
                   const _Float16* __restrict__ Vt, const float* __restrict__ Qp32,
                   float* __restrict__ Out) {
  const int lane = threadIdx.x & 31;
  const int wave = (blockIdx.x * blockDim.x + threadIdx.x) >> 5;
  const int qtiles = NQ / 32;                     // 32 wave-tiles per (b,h)
  const int b   = wave / (H * qtiles);
  const int rem = wave % (H * qtiles);
  const int h   = rem / qtiles;
  const int qbase = (rem % qtiles) * 32;
  const float scale = 0.044194173824159216f;      // 1/sqrt(512), faithful to reference
  const int l15   = lane & 15;
  const int half8 = (lane & 16) ? 8 : 0;

  const _Float16* Qb = Qp + (size_t)b * NQ * DV + h * DH;
  const _Float16* Kb = Kp + (size_t)b * NK * DV + h * DH;
  const _Float16* Vb = Vt + ((size_t)b * H + h) * DH * NK;

  v16h qreg[2][2];
#pragma unroll
  for (int qt = 0; qt < 2; ++qt)
#pragma unroll
    for (int c = 0; c < 2; ++c)
      qreg[qt][c] = load_b_f16(Qb, DV, qbase + qt * 16 + l15, c * 32, lane);

  v8f accO[2][4];
#pragma unroll
  for (int qt = 0; qt < 2; ++qt)
#pragma unroll
    for (int c = 0; c < 4; ++c)
      accO[qt][c] = (v8f){0.f,0.f,0.f,0.f,0.f,0.f,0.f,0.f};
  float mrun[2] = {-__builtin_inff(), -__builtin_inff()};
  float lrun[2] = {0.f, 0.f};

  for (int j = 0; j < NK; j += 32) {
    v8f st[2][2];
#pragma unroll
    for (int qt = 0; qt < 2; ++qt)
#pragma unroll
      for (int hf = 0; hf < 2; ++hf)
        st[qt][hf] = (v8f){0.f,0.f,0.f,0.f,0.f,0.f,0.f,0.f};
#pragma unroll
    for (int hf = 0; hf < 2; ++hf)
#pragma unroll
      for (int c = 0; c < 2; ++c) {
        v16h ka = load_a_f16(Kb, DV, j + hf * 16 + l15, c * 32, lane);
#pragma unroll
        for (int qt = 0; qt < 2; ++qt)
          st[qt][hf] = __builtin_amdgcn_wmma_f32_16x16x32_f16(false, ka, false,
                                                              qreg[qt][c], (short)0,
                                                              st[qt][hf], false, false);
      }

    // gfx1250 prefetch of the next nk tile (K rows and V rows)
    if (j + 32 < NK) {
      __builtin_prefetch(Kb + (size_t)(j + 32 + lane) * DV, 0, 1);
      __builtin_prefetch(Vb + (size_t)lane * NK + j + 32, 0, 1);
      __builtin_prefetch(Vb + (size_t)(lane + 32) * NK + j + 32, 0, 1);
    }

    v16h pa[2];
#pragma unroll
    for (int qt = 0; qt < 2; ++qt) {
      float s0[8], s1[8];
      float tmax = -__builtin_inff();
#pragma unroll
      for (int r = 0; r < 8; ++r) {
        s0[r] = st[qt][0][r] * scale;
        s1[r] = st[qt][1][r] * scale;
        tmax = fmaxf(tmax, fmaxf(s0[r], s1[r]));
      }
      tmax = fmaxf(tmax, __shfl_xor(tmax, 16));
      const float mnew = fmaxf(mrun[qt], tmax);
      const float corr = __expf(mrun[qt] - mnew);
      float psum = 0.f;
#pragma unroll
      for (int r = 0; r < 8; ++r) {
        s0[r] = __expf(s0[r] - mnew);
        s1[r] = __expf(s1[r] - mnew);
        psum += s0[r] + s1[r];
        pa[qt][r]     = (_Float16)s0[r];
        pa[qt][r + 8] = (_Float16)s1[r];
      }
      lrun[qt] = lrun[qt] * corr + psum + __shfl_xor(psum, 16);
      mrun[qt] = mnew;
#pragma unroll
      for (int r = 0; r < 8; ++r) {
        const float cr = __shfl(corr, r + half8);
#pragma unroll
        for (int c = 0; c < 4; ++c) accO[qt][c][r] *= cr;
      }
    }

#pragma unroll
    for (int c = 0; c < 4; ++c) {
      v16h vb = load_b_f16(Vb, NK, c * 16 + l15, j, lane);
#pragma unroll
      for (int qt = 0; qt < 2; ++qt)
        accO[qt][c] = __builtin_amdgcn_wmma_f32_16x16x32_f16(false, pa[qt], false, vb,
                                                             (short)0, accO[qt][c],
                                                             false, false);
    }
  }

  const int n = l15;
#pragma unroll
  for (int qt = 0; qt < 2; ++qt) {
    float il[8];
#pragma unroll
    for (int r = 0; r < 8; ++r) il[r] = 1.f / __shfl(lrun[qt], r + half8);
#pragma unroll
    for (int c = 0; c < 4; ++c)
#pragma unroll
      for (int r = 0; r < 8; ++r) {
        const int row = qbase + qt * 16 + half8 + r;
        const size_t ix = ((size_t)b * NQ + row) * DV + h * DH + c * 16 + n;
        Out[ix] = Qp32[ix] + accO[qt][c][r] * il[r];
      }
  }
}

// ---------------------------------------------------------------- kernel 4: layernorm
__global__ __launch_bounds__(256)
void mab_layernorm(const float* __restrict__ X, const float* __restrict__ g,
                   const float* __restrict__ bta,
                   float* __restrict__ Y32, _Float16* __restrict__ Y16) {
  __shared__ float red[256];
  const int tid = threadIdx.x;
  const size_t row = blockIdx.x;
  const float* x = X + row * DV;
  const float v0 = x[tid], v1 = x[tid + 256];
  red[tid] = v0 + v1;
  __syncthreads();
  for (int off = 128; off > 0; off >>= 1) {
    if (tid < off) red[tid] += red[tid + off];
    __syncthreads();
  }
  const float mean = red[0] * (1.f / DV);
  __syncthreads();
  const float d0 = v0 - mean, d1 = v1 - mean;
  red[tid] = d0 * d0 + d1 * d1;
  __syncthreads();
  for (int off = 128; off > 0; off >>= 1) {
    if (tid < off) red[tid] += red[tid + off];
    __syncthreads();
  }
  const float rstd = rsqrtf(red[0] * (1.f / DV) + 1e-5f);
  const float y0 = d0 * rstd * g[tid] + bta[tid];
  const float y1 = d1 * rstd * g[tid + 256] + bta[tid + 256];
  Y32[row * DV + tid] = y0;
  Y32[row * DV + tid + 256] = y1;
  if (Y16) {
    Y16[row * DV + tid] = (_Float16)y0;
    Y16[row * DV + tid + 256] = (_Float16)y1;
  }
}

// ---------------------------------------------------------------- kernel 5: out proj + relu + residual
__global__ __launch_bounds__(256)
void mab_out_gemm(const _Float16* __restrict__ A, const _Float16* __restrict__ W,
                  const float* __restrict__ bias, const float* __restrict__ Res,
                  float* __restrict__ Out, int M) {
  const int lane = threadIdx.x & 31;
  const int wave = (blockIdx.x * blockDim.x + threadIdx.x) >> 5;
  const int ngrp = DV / 64;
  const int tm   = (wave / ngrp) * 32;
  const int tng  = (wave % ngrp) * 64;
  if (tm >= M) return;
  const int l15 = lane & 15;

  v8f acc[2][4];
#pragma unroll
  for (int mt = 0; mt < 2; ++mt)
#pragma unroll
    for (int t = 0; t < 4; ++t) acc[mt][t] = (v8f){0.f,0.f,0.f,0.f,0.f,0.f,0.f,0.f};

  for (int k = 0; k < DV; k += 32) {
    v16h a0 = load_a_f16(A, DV, tm + l15,      k, lane);
    v16h a1 = load_a_f16(A, DV, tm + 16 + l15, k, lane);
    v16h bm[4];
#pragma unroll
    for (int t = 0; t < 4; ++t) bm[t] = load_b_f16(W, DV, tng + t * 16 + l15, k, lane);
#pragma unroll
    for (int t = 0; t < 4; ++t) {
      acc[0][t] = __builtin_amdgcn_wmma_f32_16x16x32_f16(false, a0, false, bm[t],
                                                         (short)0, acc[0][t], false, false);
      acc[1][t] = __builtin_amdgcn_wmma_f32_16x16x32_f16(false, a1, false, bm[t],
                                                         (short)0, acc[1][t], false, false);
    }
  }
  const int mofs = (lane & 16) ? 8 : 0;
#pragma unroll
  for (int mt = 0; mt < 2; ++mt)
#pragma unroll
    for (int t = 0; t < 4; ++t) {
      const int col = tng + t * 16 + l15;
      const float bv = bias[col];
#pragma unroll
      for (int r = 0; r < 8; ++r) {
        const size_t ix = (size_t)(tm + mt * 16 + mofs + r) * DV + col;
        Out[ix] = fmaxf(acc[mt][t][r] + bv, 0.f) + Res[ix];
      }
    }
}

// ---------------------------------------------------------------- host launcher
extern "C" void kernel_launch(void* const* d_in, const int* in_sizes, int n_in,
                              void* d_out, int out_size, void* d_ws, size_t ws_size,
                              hipStream_t stream) {
  const float* Q  = (const float*)d_in[0];
  const float* K  = (const float*)d_in[1];
  const float* Wq = (const float*)d_in[2];  const float* bq = (const float*)d_in[3];
  const float* Wk = (const float*)d_in[4];  const float* bk = (const float*)d_in[5];
  const float* Wv = (const float*)d_in[6];  const float* bv = (const float*)d_in[7];
  const float* Wo = (const float*)d_in[8];  const float* bo = (const float*)d_in[9];
  const float* g0 = (const float*)d_in[10]; const float* b0 = (const float*)d_in[11];
  const float* g1 = (const float*)d_in[12]; const float* b1 = (const float*)d_in[13];
  float* out = (float*)d_out;

  const size_t elems = (size_t)BB * NQ * DV;       // 16,777,216
  const size_t welems = (size_t)DV * DV;           // 262,144
  char* ws = (char*)d_ws;
  float*    Qp32  = (float*)ws;    ws += elems * sizeof(float);     // reused as O1 f32
  _Float16* Qpb   = (_Float16*)ws; ws += elems * sizeof(_Float16);  // reused as O1 f16
  _Float16* Kpb   = (_Float16*)ws; ws += elems * sizeof(_Float16);
  _Float16* Vpb   = (_Float16*)ws; ws += elems * sizeof(_Float16);
  _Float16* Vtb   = (_Float16*)ws; ws += elems * sizeof(_Float16);
  float*    O1pre = (float*)ws;    ws += elems * sizeof(float);     // reused as R f32
  _Float16* Qh16  = (_Float16*)ws; ws += elems * sizeof(_Float16);
  _Float16* Kh16  = (_Float16*)ws; ws += elems * sizeof(_Float16);
  _Float16* Wq16  = (_Float16*)ws; ws += welems * sizeof(_Float16);
  _Float16* Wk16  = (_Float16*)ws; ws += welems * sizeof(_Float16);
  _Float16* Wv16  = (_Float16*)ws; ws += welems * sizeof(_Float16);
  _Float16* Wo16  = (_Float16*)ws; ws += welems * sizeof(_Float16);

  const int M = BB * NQ;                                  // 32768 rows
  const dim3 blk(256);
  const int cvtBig  = (int)(elems / (256 * 8));           // 8192
  const int cvtW    = (int)(welems / (256 * 8));          // 128
  const int gemmBlocks = (M / 32) * (DV / 64) / 8;        // 1024
  const int attnBlocks = (BB * H * (NQ / 32)) / 8;        // 1024
  const int trBlocks   = (int)(elems / 256);              // 65536

  // one-shot f32 -> f16 conversions (kills all converts in GEMM hot loops)
  mab_cvt_f16<<<cvtBig, blk, 0, stream>>>(Q,  Qh16, elems);
  mab_cvt_f16<<<cvtBig, blk, 0, stream>>>(K,  Kh16, elems);
  mab_cvt_f16<<<cvtW,  blk, 0, stream>>>(Wq, Wq16, welems);
  mab_cvt_f16<<<cvtW,  blk, 0, stream>>>(Wk, Wk16, welems);
  mab_cvt_f16<<<cvtW,  blk, 0, stream>>>(Wv, Wv16, welems);
  mab_cvt_f16<<<cvtW,  blk, 0, stream>>>(Wo, Wo16, welems);

  // projections (all-f16 WMMA, f32+f16 out)
  mab_proj_gemm<<<gemmBlocks, blk, 0, stream>>>(Qh16, Wq16, bq, Qp32, Qpb, M);
  mab_proj_gemm<<<gemmBlocks, blk, 0, stream>>>(Kh16, Wk16, bk, nullptr, Kpb, M);
  mab_proj_gemm<<<gemmBlocks, blk, 0, stream>>>(Kh16, Wv16, bv, nullptr, Vpb, M);
  // V -> [b,h,dh,nk] for contiguous B-operand loads in PV
  mab_transpose_v<<<trBlocks, blk, 0, stream>>>(Vpb, Vtb);
  // flash attention + residual (reads Qp32 residual, writes O1pre)
  mab_attention<<<attnBlocks, blk, 0, stream>>>(Qpb, Kpb, Vtb, Qp32, O1pre);
  // LN0: O1pre -> O1 (f32 into Qp32 buffer, f16 into Qpb buffer)
  mab_layernorm<<<M, blk, 0, stream>>>(O1pre, g0, b0, Qp32, Qpb);
  // out proj + relu + residual: R into O1pre buffer
  mab_out_gemm<<<gemmBlocks, blk, 0, stream>>>(Qpb, Wo16, bo, Qp32, O1pre, M);
  // LN1 -> final output
  mab_layernorm<<<M, blk, 0, stream>>>(O1pre, g1, b1, out, nullptr);
}
